// NodeInfoPropagator_80410377715747
// MI455X (gfx1250) — matrix operand order repaired and checked by reading, not compiled, not run on hardware
//
#include <hip/hip_runtime.h>
#include <math.h>

// ---------------------------------------------------------------------------
// NodeInfoPropagator on gfx1250 (MI455X).
// All GEMMs: v_wmma_f32_16x16x32_f16 (f32 accumulate).
// Weight slabs staged to LDS via global_load_async_to_lds_b128 (ASYNCcnt).
// GRU kernel register-blocks 2 col-tiles per wave: 12 WMMAs per A-frag pair.
// ---------------------------------------------------------------------------

#define USE_ASYNC_STAGE 1

typedef __attribute__((ext_vector_type(16))) _Float16 v16h;
typedef __attribute__((ext_vector_type(8)))  _Float16 v8h;
typedef __attribute__((ext_vector_type(4)))  _Float16 v4h;
typedef __attribute__((ext_vector_type(8)))  float    v8f;

#define PDIM 256

__device__ __forceinline__ v16h cat8(v8h lo, v8h hi) {
  return __builtin_shufflevector(lo, hi, 0, 1, 2, 3, 4, 5, 6, 7,
                                 8, 9, 10, 11, 12, 13, 14, 15);
}

__device__ __forceinline__ v8f wmma_f16(v16h a, v16h b, v8f c) {
  return __builtin_amdgcn_wmma_f32_16x16x32_f16(false, a, false, b, (short)0, c, false, false);
}

// A fragment (16x32 f16, ISA 7.12.2): lane holds row lane&15,
// kBase = (lane>=16 ? 8 : 0); halves [kb..kb+7] and [kb+16..kb+23].
__device__ __forceinline__ v16h load_a16(const _Float16* __restrict__ rowPtr, int k0, int hi) {
  const int kb = k0 + (hi ? 8 : 0);
  v8h lo = *(const v8h*)(rowPtr + kb);
  v8h up = *(const v8h*)(rowPtr + kb + 16);
  return cat8(lo, up);
}

// Async global->LDS 16B copy (GLOBAL_LOAD_ASYNC_TO_LDS_B128, ASYNCcnt).
// ldsPtr low 32 bits of the generic address are the LDS byte offset
// (LDS aperture truncation, ISA 10.2).
__device__ __forceinline__ void async_copy_b128(const void* ldsPtr, const void* gptr) {
#if USE_ASYNC_STAGE
  unsigned lds = (unsigned)(unsigned long long)ldsPtr;
  unsigned long long ga = (unsigned long long)gptr;
  asm volatile("global_load_async_to_lds_b128 %0, %1, off"
               :: "v"(lds), "v"(ga) : "memory");
#else
  *(v8h*)ldsPtr = *(const v8h*)gptr;
#endif
}

__device__ __forceinline__ void async_stage_fence() {
#if USE_ASYNC_STAGE
  asm volatile("s_wait_asynccnt 0" ::: "memory");
#endif
}

__device__ __forceinline__ float sigmoidf_fast(float x) {
  return 1.0f / (1.0f + __expf(-x));
}

// ---------------------------------------------------------------------------
// Generic f32 -> f16 convert: out[m,k] = (f16)(src[gmap?gmap[m]:m, k] * scale(m))
// Covers: weight conversion, node_infos[perm] staging, child_sum/fan staging.
// ---------------------------------------------------------------------------
__global__ __launch_bounds__(256) void convert_rows_kernel(
    const float* __restrict__ src, const int* __restrict__ gmap,
    const int* __restrict__ fanouts, _Float16* __restrict__ out,
    int Mrows, int K) {
  const long long t = (long long)blockIdx.x * blockDim.x + threadIdx.x;
  const int perRow = K >> 2;
  const long long total = (long long)Mrows * perRow;
  if (t >= total) return;
  const int m = (int)(t / perRow);
  const int s = (int)(t - (long long)m * perRow) << 2;
  const int sm = gmap ? gmap[m] : m;
  float sc = 1.0f;
  if (fanouts) {
    int f = fanouts[m];
    sc = 1.0f / (float)(f < 1 ? 1 : f);
  }
  const float4 v = *(const float4*)(src + (size_t)sm * K + s);
  v4h o;
  o[0] = (_Float16)(v.x * sc); o[1] = (_Float16)(v.y * sc);
  o[2] = (_Float16)(v.z * sc); o[3] = (_Float16)(v.w * sc);
  *(v4h*)(out + (size_t)m * K + s) = o;
}

// ---------------------------------------------------------------------------
// Tiled GEMM, f16 operands, f32 accumulate:
//   Out[m,n] = sum_k A16[gather?gather[m]:m, k] * W16[n, k]
//              + (Cinit ? Cinit[m,n] : 0) + bias[n]
// Block = 8 waves, one col-tile, 8 row-tiles; B slab (16 x K) async-staged.
// ---------------------------------------------------------------------------
__global__ __launch_bounds__(256) void gemm16_kernel(
    const _Float16* __restrict__ A16, const int* __restrict__ gather,
    const _Float16* __restrict__ W16, const float* __restrict__ bias,
    const float* __restrict__ Cinit,
    _Float16* __restrict__ Out16, float* __restrict__ Out32,
    int M, int K, int Ncols) {
  __shared__ _Float16 lsB[16 * 512];           // supports K up to 512 (16 KB)
  const int lane = threadIdx.x & 31;
  const int wave = threadIdx.x >> 5;
  const int colTiles = Ncols >> 4;
  const int ct = blockIdx.x % colTiles;
  const int rg = blockIdx.x / colTiles;
  const int colBase = ct << 4;

  // async stage of B block (16 rows of W, full K) into LDS
  const int kChunks = K >> 3;                   // v8h chunks per row
  for (int i = threadIdx.x; i < 16 * kChunks; i += blockDim.x) {
    const int n = i / kChunks;
    const int koff = (i - n * kChunks) << 3;
    async_copy_b128(lsB + n * K + koff, W16 + (size_t)(colBase + n) * K + koff);
  }
  async_stage_fence();
  __syncthreads();

  const int rt = rg * 8 + wave;
  if (rt >= (M >> 4)) return;
  const int rowBase = rt << 4;
  const int hi = (lane & 16) ? 1 : 0;
  const int mrow = rowBase + (lane & 15);
  const int arow = gather ? gather[mrow] : mrow;
  const _Float16* aRow = A16 + (size_t)arow * K;
  const int col = colBase + (lane & 15);
  // B frag (ISA layout): lane holds col n=lane&15, kb=(lane>=16?8:0),
  // elements -> k0+kb+[0..7] and k0+kb+16+[0..7] in LDS.
  const _Float16* bPtr = lsB + (lane & 15) * K + (hi ? 8 : 0);

  v8f acc;
  if (Cinit) {
#pragma unroll
    for (int r = 0; r < 8; ++r)
      acc[r] = Cinit[(size_t)(rowBase + r + hi * 8) * Ncols + col];
  } else {
#pragma unroll
    for (int r = 0; r < 8; ++r) acc[r] = 0.0f;
  }

  for (int k0 = 0; k0 < K; k0 += 32) {
    v16h a = load_a16(aRow, k0, hi);
    v16h b = cat8(*(const v8h*)(bPtr + k0), *(const v8h*)(bPtr + k0 + 16));
    acc = wmma_f16(a, b, acc);
  }

  const float bv = bias[col];
#pragma unroll
  for (int r = 0; r < 8; ++r) {
    const size_t idx = (size_t)(rowBase + r + hi * 8) * Ncols + col;
    const float v = acc[r] + bv;
    if (Out16) Out16[idx] = (_Float16)v;
    if (Out32) Out32[idx] = v;
  }
}

// ---------------------------------------------------------------------------
// Edge scatter-add: CS[dst[e], :] += (f32)H16[src[e], :]
// ---------------------------------------------------------------------------
__global__ __launch_bounds__(256) void child_scatter_kernel(
    const _Float16* __restrict__ H16, const int* __restrict__ src,
    const int* __restrict__ dst, float* __restrict__ CS, int E) {
  const int t = blockIdx.x * blockDim.x + threadIdx.x;
  const int segs = PDIM / 8;                    // 32 v8h segments per row
  const int e = t / segs;
  if (e >= E) return;
  const int s = (t - e * segs) * 8;
  const v8h v = *(const v8h*)(H16 + (size_t)src[e] * PDIM + s);
  float* base = CS + (size_t)dst[e] * PDIM + s;
#pragma unroll
  for (int i = 0; i < 8; ++i)
    __hip_atomic_fetch_add(base + i, (float)v[i], __ATOMIC_RELAXED, __HIP_MEMORY_SCOPE_AGENT);
}

// ---------------------------------------------------------------------------
// Fused GRU step. Block = 8 waves, TWO col-tiles (32 cols), 8 row-tiles.
// 6 weight slabs (Wih r/z/n, Whh r/z/n; 32 x 256 each) async-staged (96 KB).
// Per k-step: 4 global b128 A loads feed 12 WMMAs (12 f32 accumulators).
// ---------------------------------------------------------------------------
__global__ __launch_bounds__(256) void gru16_kernel(
    const _Float16* __restrict__ H16, const _Float16* __restrict__ N16,
    const _Float16* __restrict__ Wih16, const float* __restrict__ bih,
    const _Float16* __restrict__ Whh16, const float* __restrict__ bhh,
    _Float16* __restrict__ Out16, float* __restrict__ Out32,
    const int* __restrict__ scatterRows, int M) {
  __shared__ _Float16 lsB[6 * 32 * PDIM];       // 96 KB (of 320 KB WGP LDS)
  const int K = PDIM;
  const int lane = threadIdx.x & 31;
  const int wave = threadIdx.x >> 5;
  const int colGroups = PDIM >> 5;              // 8 groups of 32 cols
  const int cg = blockIdx.x % colGroups;
  const int rg = blockIdx.x / colGroups;
  const int colBase = cg << 5;

  // stage 6 weight slabs (32 cols x 256 k each):
  // mat 0..2 = Wih rows {0,256,512}+colBase ; mat 3..5 = Whh same
  const int kChunks = K >> 3;                   // 32
  const int chunksPerMat = 32 * kChunks;        // 1024
  for (int i = threadIdx.x; i < 6 * chunksPerMat; i += blockDim.x) {
    const int mat = i / chunksPerMat;
    const int r = i - mat * chunksPerMat;
    const int n = r / kChunks;
    const int koff = (r - n * kChunks) << 3;
    const _Float16* srcW = (mat < 3) ? Wih16 : Whh16;
    const int nrow = (mat % 3) * PDIM + colBase + n;
    async_copy_b128(lsB + (size_t)(mat * 32 + n) * K + koff,
                    srcW + (size_t)nrow * K + koff);
  }
  async_stage_fence();
  __syncthreads();

  const int rt = rg * 8 + wave;
  if (rt >= (M >> 4)) return;
  const int rowBase = rt << 4;
  const int hi = (lane & 16) ? 1 : 0;
  const _Float16* hRow = H16 + (size_t)(rowBase + (lane & 15)) * K;
  const _Float16* nRow = N16 + (size_t)(rowBase + (lane & 15)) * K;
  const int khalf = (hi ? 8 : 0);
  const int nIdx = lane & 15;

  // acc[s*6 + {0:ir,1:iz,2:in,3:hr,4:hz,5:hn}] for subtile s in {0,1}
  v8f acc[12];
#pragma unroll
  for (int i = 0; i < 12; ++i) {
#pragma unroll
    for (int r = 0; r < 8; ++r) acc[i][r] = 0.0f;
  }

  for (int k0 = 0; k0 < K; k0 += 32) {
    v16h ah = load_a16(hRow, k0, hi);
    v16h an = load_a16(nRow, k0, hi);
#pragma unroll
    for (int s = 0; s < 2; ++s) {
#pragma unroll
      for (int m = 0; m < 3; ++m) {
        const _Float16* bp = lsB + (size_t)(m * 32 + s * 16 + nIdx) * K + khalf + k0;
        acc[s * 6 + m] = wmma_f16(ah, cat8(*(const v8h*)bp, *(const v8h*)(bp + 16)),
                                  acc[s * 6 + m]);
      }
#pragma unroll
      for (int m = 0; m < 3; ++m) {
        const _Float16* bp = lsB + (size_t)((m + 3) * 32 + s * 16 + nIdx) * K + khalf + k0;
        acc[s * 6 + 3 + m] = wmma_f16(an, cat8(*(const v8h*)bp, *(const v8h*)(bp + 16)),
                                      acc[s * 6 + 3 + m]);
      }
    }
  }

#pragma unroll
  for (int s = 0; s < 2; ++s) {
    const int col = colBase + s * 16 + nIdx;
    const float bir = bih[col], biz = bih[col + PDIM], bin = bih[col + 2 * PDIM];
    const float bhr = bhh[col], bhz = bhh[col + PDIM], bhn = bhh[col + 2 * PDIM];
#pragma unroll
    for (int r = 0; r < 8; ++r) {
      const int row = rowBase + r + hi * 8;
      const float nbrv = (float)N16[(size_t)row * PDIM + col];
      const float rg_ = sigmoidf_fast(acc[s * 6 + 0][r] + bir + acc[s * 6 + 3][r] + bhr);
      const float zg  = sigmoidf_fast(acc[s * 6 + 1][r] + biz + acc[s * 6 + 4][r] + bhz);
      const float ng  = tanhf(acc[s * 6 + 2][r] + bin + rg_ * (acc[s * 6 + 5][r] + bhn));
      const float hv  = (1.0f - zg) * ng + zg * nbrv;
      if (Out32) {
        const int orow = scatterRows ? scatterRows[row] : row;
        Out32[(size_t)orow * PDIM + col] = hv;
      } else {
        Out16[(size_t)row * PDIM + col] = (_Float16)hv;
      }
    }
  }
}

// ---------------------------------------------------------------------------
extern "C" void kernel_launch(void* const* d_in, const int* in_sizes, int n_in,
                              void* d_out, int out_size, void* d_ws, size_t ws_size,
                              hipStream_t stream) {
  const float* node_infos = (const float*)d_in[0];
  const float* W_resize   = (const float*)d_in[1];
  const float* b_resize   = (const float*)d_in[2];
  const float* W_parent   = (const float*)d_in[3];
  const float* b_parent   = (const float*)d_in[4];
  const float* W_nbr      = (const float*)d_in[5];
  const float* b_nbr      = (const float*)d_in[6];
  const float* W_ih       = (const float*)d_in[7];
  const float* b_ih       = (const float*)d_in[8];
  const float* W_hh       = (const float*)d_in[9];
  const float* b_hh       = (const float*)d_in[10];
  const int*   perm       = (const int*)d_in[11];
  const int*   parent_pos = (const int*)d_in[13];
  const int*   child_src  = (const int*)d_in[14];
  const int*   child_dst  = (const int*)d_in[15];
  const int*   fanouts    = (const int*)d_in[16];

  const int M   = in_sizes[11];                 // 131072
  const int P   = in_sizes[2];                  // 256
  const int EIN = in_sizes[1] / P;              // 512
  const int E   = in_sizes[14];                 // 130944
  const int DEPTH_C = 3;  // reference module constant DEPTH=3

  const size_t mp = (size_t)M * P;              // 33.5M elems

  // Workspace layout:
  //  R0: max(ni16 (M*EIN*2), cs32 (mp*4)) = 128MB (ni16 dead before loop)
  //  R1/R2: h16 ping/pong (mp*2 each); R3: pt32 (mp*4);
  //  R4: cs16 (mp*2); R5: nbr16 (mp*2); R6: f16 weights (~1.3MB)
  char* base = (char*)d_ws;
  _Float16* ni16 = (_Float16*)base;             // M x EIN (pre-loop only)
  float*    cs32 = (float*)base;                // M x P   (loop only)
  base += mp * 4;
  _Float16* h16a = (_Float16*)base; base += mp * 2;
  _Float16* h16b = (_Float16*)base; base += mp * 2;
  float*    pt32 = (float*)base;    base += mp * 4;
  _Float16* cs16 = (_Float16*)base; base += mp * 2;
  _Float16* nb16 = (_Float16*)base; base += mp * 2;
  _Float16* wr16  = (_Float16*)base; base += (size_t)P * EIN * 2;
  _Float16* wp16  = (_Float16*)base; base += (size_t)P * P * 2;
  _Float16* wn16  = (_Float16*)base; base += (size_t)P * P * 2;
  _Float16* wih16 = (_Float16*)base; base += (size_t)3 * P * P * 2;
  _Float16* whh16 = (_Float16*)base;

  const dim3 blk(256);
  auto cvblocks = [](long long elems) { return (int)((elems / 4 + 255) / 256); };

  // ---- one-time f16 staging ----
  convert_rows_kernel<<<cvblocks((long long)P * EIN), blk, 0, stream>>>(
      W_resize, nullptr, nullptr, wr16, P, EIN);
  convert_rows_kernel<<<cvblocks((long long)P * P), blk, 0, stream>>>(
      W_parent, nullptr, nullptr, wp16, P, P);
  convert_rows_kernel<<<cvblocks((long long)P * P), blk, 0, stream>>>(
      W_nbr, nullptr, nullptr, wn16, P, P);
  convert_rows_kernel<<<cvblocks((long long)3 * P * P), blk, 0, stream>>>(
      W_ih, nullptr, nullptr, wih16, 3 * P, P);
  convert_rows_kernel<<<cvblocks((long long)3 * P * P), blk, 0, stream>>>(
      W_hh, nullptr, nullptr, whh16, 3 * P, P);
  // ni16[i] = (f16) node_infos[perm[i]]  (perm fused into staging)
  convert_rows_kernel<<<cvblocks((long long)M * EIN), blk, 0, stream>>>(
      node_infos, perm, nullptr, ni16, M, EIN);

  const int gblocks  = (P / 16) * (M / 128);    // 16384 (gemm16)
  const int grblocks = (P / 32) * (M / 128);    // 8192  (gru16)

  // h = ni16 @ W_resize^T + b_resize            -> h16a
  gemm16_kernel<<<gblocks, blk, 0, stream>>>(ni16, nullptr, wr16, b_resize,
                                             nullptr, h16a, nullptr, M, EIN, P);
  // parent_term = h[parent_pos] @ W_parent^T + b -> pt32 (f32 C-init source)
  gemm16_kernel<<<gblocks, blk, 0, stream>>>(h16a, parent_pos, wp16, b_parent,
                                             nullptr, nullptr, pt32, M, P, P);

  _Float16* hCur = h16a;
  _Float16* hNext = h16b;
  const long long sthreads = (long long)E * (P / 8);
  const int sblocks = (int)((sthreads + 255) / 256);

  for (int it = 0; it < DEPTH_C; ++it) {
    hipMemsetAsync(cs32, 0, mp * sizeof(float), stream);
    child_scatter_kernel<<<sblocks, blk, 0, stream>>>(hCur, child_src, child_dst, cs32, E);
    // cs16 = (f16)(child_sum / fan)
    convert_rows_kernel<<<cvblocks((long long)M * P), blk, 0, stream>>>(
        cs32, nullptr, fanouts, cs16, M, P);
    // nbr = parent_term + cs16 @ W_nbr^T + b_nbr -> nb16
    gemm16_kernel<<<gblocks, blk, 0, stream>>>(cs16, nullptr, wn16, b_nbr,
                                               pt32, nb16, nullptr, M, P, P);
    const bool last = (it == DEPTH_C - 1);
    gru16_kernel<<<grblocks, blk, 0, stream>>>(
        hCur, nb16, wih16, b_ih, whh16, b_hh,
        last ? nullptr : hNext, last ? (float*)d_out : nullptr,
        last ? perm : nullptr, M);
    _Float16* tmp = hCur; hCur = hNext; hNext = tmp;
  }
}
